// HashAttention_52553219834304
// MI455X (gfx1250) — matrix operand order repaired and checked by reading, not compiled, not run on hardware
//
#include <hip/hip_runtime.h>
#include <cstdint>

// ---------------- constants ----------------
#define NB 4
#define NLEV 16
#define TDIM 8192
#define CH 32
#define SDIM 512
#define RES 256
#define HRES 128
#define PADW 258           // 256 + 2 halo
#define PADA (PADW*PADW)   // 66564

__device__ __constant__ int c_res[NLEV] = {16,19,23,27,33,40,48,58,70,84,101,122,147,176,212,256};

typedef _Float16 v16h __attribute__((ext_vector_type(16)));
typedef _Float16 v8h  __attribute__((ext_vector_type(8)));
typedef _Float16 v2h  __attribute__((ext_vector_type(2)));
typedef float    v8f  __attribute__((ext_vector_type(8)));

__device__ static inline v8f wmma_f16f32(v16h a, v16h b, v8f c) {
  return __builtin_amdgcn_wmma_f32_16x16x32_f16(false, a, false, b, (short)0, c, false, false);
}

__device__ static inline v16h join8(v8h lo, v8h hi) {
  return __builtin_shufflevector(lo, hi, 0,1,2,3,4,5,6,7,8,9,10,11,12,13,14,15);
}

// ---------------- async global->LDS (gfx1250) with fallback ----------------
#if defined(__HIP_DEVICE_COMPILE__) && __has_builtin(__builtin_amdgcn_global_load_async_to_lds_b128)
#define HAVE_ASYNC 1
#else
#define HAVE_ASYNC 0
#endif

#if HAVE_ASYNC
typedef int v4i_t __attribute__((vector_size(16)));
typedef __attribute__((address_space(1))) v4i_t* gas1_v4i;   // global int4*
typedef __attribute__((address_space(3))) v4i_t* las3_v4i;   // LDS int4*
__device__ static inline void async_copy_b128(const float* g, float* l) {
  __builtin_amdgcn_global_load_async_to_lds_b128(
      (gas1_v4i)(uintptr_t)g,
      (las3_v4i)(unsigned int)(uintptr_t)l,
      0, 0);
}
#endif

__device__ static inline void wait_async_all() {
#if HAVE_ASYNC
#if __has_builtin(__builtin_amdgcn_s_wait_asynccnt)
  __builtin_amdgcn_s_wait_asynccnt(0);
#else
  asm volatile("s_wait_asynccnt 0" ::: "memory");
#endif
#endif
}

// ---------------- tiny utility kernels ----------------
__global__ void k_zero32(unsigned int* p, long n) {
  long i = (long)blockIdx.x * blockDim.x + threadIdx.x;
  if (i < n) p[i] = 0u;
}

// styles: set 0 -> conv style, 1 -> k style, 2 -> o style.  grid=3, block=128
__global__ void k_styles(const float* s, const float* aw, const float* ab,
                         const float* kaw, const float* kab,
                         const float* oaw, const float* oab,
                         float* sst, float* kst, float* ost) {
  int set = blockIdx.x, t = threadIdx.x;
  int b = t >> 5, o = t & 31;
  const float* W  = (set == 0) ? aw : (set == 1) ? kaw : oaw;
  const float* Bb = (set == 0) ? ab : (set == 1) ? kab : oab;
  float*       O  = (set == 0) ? sst : (set == 1) ? kst : ost;
  const float* sv = s + (size_t)b * SDIM;
  const float* wv = W + (size_t)o * SDIM;
  float acc = 0.f;
  for (int d = 0; d < SDIM; ++d) acc += sv[d] * wv[d];
  O[b * CH + o] = acc * 0.04419417382415922f + Bb[o];   // 1/sqrt(512)
}

// modulate+demodulate weights.  Conv weights stored K-contiguous per output
// channel: wT[b][co][k], k = tap*32 + ci  (tap = kh*3+kw) -> each WMMA B
// fragment is 16 consecutive f16 (one 32B vector load).  Same for kB/oB.
// grid = NB*CH (one block per (b,co)), block = 320
__global__ void __launch_bounds__(320) k_modprep(
    const float* weight, const float* kw, const float* ow,
    const float* sst, const float* kst, const float* ost,
    _Float16* wT, _Float16* kB, _Float16* oB) {
  __shared__ float sarr[512];
  int blk = blockIdx.x, b = blk >> 5, co = blk & 31, t = threadIdx.x;
  float wmv = 0.f;
  int ci = 0, rr = 0;
  if (t < 288) {
    ci = t / 9; rr = t - ci * 9;
    wmv = weight[(size_t)co * 288 + ci * 9 + rr] * sst[b * CH + ci];
  }
  sarr[t] = (t < 288) ? wmv * wmv : 0.f;
  if (t < 192) sarr[320 + t] = 0.f;
  __syncthreads();
  for (int st = 256; st > 0; st >>= 1) { if (t < st) sarr[t] += sarr[t + st]; __syncthreads(); }
  float d = rsqrtf(sarr[0] + 1e-8f);
  if (t < 288) wT[((size_t)(b * CH + co)) * 288 + (rr * 32 + ci)] = (_Float16)(wmv * d);
  if (t < 32) {  // one wave: modulated linear weights, demod over input dim
    float kv = kw[co * CH + t] * kst[b * CH + t];
    float ks = kv * kv;
    for (int m = 16; m >= 1; m >>= 1) ks += __shfl_xor(ks, m, 32);
    kB[((size_t)(b * CH + co)) * CH + t] = (_Float16)(kv * rsqrtf(ks + 1e-8f));
    float ov = ow[co * CH + t] * ost[b * CH + t];
    float os = ov * ov;
    for (int m = 16; m >= 1; m >>= 1) os += __shfl_xor(os, m, 32);
    oB[((size_t)(b * CH + co)) * CH + t] = (_Float16)(ov * rsqrtf(os + 1e-8f));
  }
}

// ---------------- hash retrieve: table staged in LDS via async DMA ----------
// xh layout is pixel-major, channel-contiguous: xh[b][row][col][ci] so the
// conv A-fragment K-run (ci) is memory-contiguous.
// grid = NB*NLEV, block = 1024, dynamic LDS = 64KB
__global__ void __launch_bounds__(1024) k_retrieve(const float* inputs, _Float16* xh) {
  extern __shared__ float tbl[];               // 16384 floats = one table
  int bl = blockIdx.x, b = bl >> 4, l = bl & 15, tid = threadIdx.x;
  const float* src = inputs + (size_t)bl * (TDIM * 2);
#if HAVE_ASYNC
  for (int i = tid * 4; i < TDIM * 2; i += 1024 * 4) async_copy_b128(src + i, tbl + i);
  wait_async_all();
  __syncthreads();
#else
  for (int i = tid * 4; i < TDIM * 2; i += 1024 * 4) *(float4*)(tbl + i) = *(const float4*)(src + i);
  __syncthreads();
#endif
  int   rm = c_res[l] - 1;
  float sc = (float)rm * (1.0f / RES);
  _Float16* xhb = xh + (size_t)b * PADA * CH + 2 * l;
  for (int p = tid; p < RES * RES; p += 1024) {
    int x = p & 255, y = p >> 8;
    float fx = ((float)x + 0.5f) * sc, fy = ((float)y + 0.5f) * sc;
    float flx = floorf(fx), fly = floorf(fy);
    int xi0 = (int)flx, yi0 = (int)fly;
    float tx = fx - flx, ty = fy - fly;
    int xi1 = min(xi0 + 1, rm), yi1 = min(yi0 + 1, rm);
    float w00 = (1.f - tx) * (1.f - ty), w10 = tx * (1.f - ty);
    float w01 = (1.f - tx) * ty,         w11 = tx * ty;
    unsigned h0 = (unsigned)yi0 * 2654435761u, h1 = (unsigned)yi1 * 2654435761u;
    unsigned i00 = ((unsigned)xi0 ^ h0) & 8191u, i10 = ((unsigned)xi1 ^ h0) & 8191u;
    unsigned i01 = ((unsigned)xi0 ^ h1) & 8191u, i11 = ((unsigned)xi1 ^ h1) & 8191u;
    float a0 = tbl[2*i00]   * w00 + tbl[2*i10]   * w10 + tbl[2*i01]   * w01 + tbl[2*i11]   * w11;
    float a1 = tbl[2*i00+1] * w00 + tbl[2*i10+1] * w10 + tbl[2*i01+1] * w01 + tbl[2*i11+1] * w11;
    v2h pk = { (_Float16)a0, (_Float16)a1 };     // channels 2l, 2l+1 adjacent
    *(v2h*)(xhb + ((size_t)(y + 1) * PADW + (x + 1)) * CH) = pk;
  }
}

// ---------------- modulated 3x3 conv as WMMA implicit GEMM ------------------
// grid = (512, NB), block = 256 (8 waves); wave tile = 16 px x 32 co, K=288
__global__ void __launch_bounds__(256) k_conv(const _Float16* xh, const _Float16* wT, float* y) {
  int b = blockIdx.y;
  int wv = threadIdx.x >> 5, lane = threadIdx.x & 31;
  int m = lane & 15, hk = lane >> 4;
  int pb = blockIdx.x * 128 + wv * 16;
  int py = pb >> 8, px = pb & 255;
  const _Float16* xb = xh + (size_t)b * PADA * CH;
  const _Float16* wb = wT + (size_t)b * CH * 288;
  const _Float16* wr0 = wb + (size_t)m * 288 + hk * 16;          // co = m
  const _Float16* wr1 = wb + (size_t)(m + 16) * 288 + hk * 16;   // co = m+16
  v8f acc0 = {0,0,0,0,0,0,0,0}, acc1 = {0,0,0,0,0,0,0,0};
  for (int kc = 0; kc < 9; ++kc) {               // tap index == k-chunk
    int kh = kc / 3, kwd = kc - 3 * kh;
    // A fragment: channels contiguous at this pixel -> two 16B vector loads
    const _Float16* axp = xb + ((size_t)(py + kh) * PADW + (px + m + kwd)) * CH + hk * 16;
    v16h a = join8(*(const v8h*)axp, *(const v8h*)(axp + 8));
    // B fragments: 16 consecutive f16 per lane (K-contiguous weight layout)
    v16h b0 = *(const v16h*)(wr0 + kc * 32);
    v16h b1 = *(const v16h*)(wr1 + kc * 32);
    acc0 = wmma_f16f32(a, b0, acc0);
    acc1 = wmma_f16f32(a, b1, acc1);
  }
  float* yb = y + (size_t)b * CH * (RES * RES);
#pragma unroll
  for (int r = 0; r < 8; ++r) {                   // D layout: M = r + hk*8, N = m
    int pix = pb + r + hk * 8;
    yb[(size_t)m * (RES * RES) + pix]        = acc0[r];
    yb[(size_t)(m + 16) * (RES * RES) + pix] = acc1[r];
  }
}

// ---------------- blur 4x4 stride-2 + bias + leaky-relu --------------------
__global__ void __launch_bounds__(256) k_blur(const float* y, const float* bias,
                                              float* tokf, _Float16* tokh) {
  int gid = blockIdx.x * 256 + threadIdx.x;        // NB*CH*HRES*HRES
  int b = gid >> 19, rem = gid & 524287;
  int co = rem >> 14, p = rem & 16383;
  int oy = p >> 7, ox = p & 127;
  const float* yb = y + ((size_t)b * CH + co) * (RES * RES);
  const float w4[4] = {1.f, 3.f, 3.f, 1.f};
  float acc = 0.f;
#pragma unroll
  for (int i = 0; i < 4; ++i) {
    int iy = 2 * oy + i - 1;
    if ((unsigned)iy < (unsigned)RES) {
#pragma unroll
      for (int j = 0; j < 4; ++j) {
        int ix = 2 * ox + j - 1;
        if ((unsigned)ix < (unsigned)RES) acc += yb[iy * RES + ix] * w4[i] * w4[j];
      }
    }
  }
  acc = acc * (1.f / 64.f) + bias[co];
  acc = (acc >= 0.f ? acc : 0.2f * acc) * 1.41421356237f;
  size_t o = ((size_t)b * 16384 + p) * CH + co;
  tokf[o] = acc;
  tokh[o] = (_Float16)acc;
}

// ---------------- MHA (identity attention) = 2 WMMA linears + LN -----------
// grid = (128, NB), block = 256
__global__ void __launch_bounds__(256) k_mha(const float* tokf, const _Float16* tokh,
                                             const _Float16* kB, const _Float16* oB,
                                             float* tokout) {
  __shared__ _Float16 ldsK[128 * CH];
  __shared__ float    ldsF[128 * CH];
  int b = blockIdx.y;
  int wv = threadIdx.x >> 5, lane = threadIdx.x & 31;
  int m = lane & 15, hk = lane >> 4;
  int tb = blockIdx.x * 128 + wv * 16;
  const _Float16* th = tokh + (size_t)b * 16384 * CH;
  const _Float16* kb = kB + (size_t)b * CH * CH;
  const _Float16* ob = oB + (size_t)b * CH * CH;
  // A fragment: rows are K-contiguous -> two 16B vector loads
  const _Float16* ap = th + (size_t)(tb + m) * CH + hk * 16;
  v16h a = join8(*(const v8h*)ap, *(const v8h*)(ap + 8));
  // B fragments: K-contiguous weight layout -> one 32B vector load each
  v16h b0 = *(const v16h*)(kb + (size_t)m * CH + hk * 16);
  v16h b1 = *(const v16h*)(kb + (size_t)(m + 16) * CH + hk * 16);
  v8f acc0 = {0,0,0,0,0,0,0,0}, acc1 = {0,0,0,0,0,0,0,0};
  acc0 = wmma_f16f32(a, b0, acc0);
  acc1 = wmma_f16f32(a, b1, acc1);
#pragma unroll
  for (int r = 0; r < 8; ++r) {                    // kqv -> LDS (re-layout for A)
    int tl = wv * 16 + r + hk * 8;
    ldsK[tl * CH + m]      = (_Float16)acc0[r];
    ldsK[tl * CH + m + 16] = (_Float16)acc1[r];
  }
  __syncthreads();
  const _Float16* a2p = ldsK + (wv * 16 + m) * CH + hk * 16;
  v16h a2 = join8(*(const v8h*)a2p, *(const v8h*)(a2p + 8));
  v16h c0 = *(const v16h*)(ob + (size_t)m * CH + hk * 16);
  v16h c1 = *(const v16h*)(ob + (size_t)(m + 16) * CH + hk * 16);
  v8f acc2 = {0,0,0,0,0,0,0,0}, acc3 = {0,0,0,0,0,0,0,0};
  acc2 = wmma_f16f32(a2, c0, acc2);
  acc3 = wmma_f16f32(a2, c1, acc3);
  const float* tf = tokf + (size_t)b * 16384 * CH;
#pragma unroll
  for (int r = 0; r < 8; ++r) {                    // + residual -> LDS
    int tl = wv * 16 + r + hk * 8;
    int tg = tb + r + hk * 8;
    ldsF[tl * CH + m]      = acc2[r] + tf[(size_t)tg * CH + m];
    ldsF[tl * CH + m + 16] = acc3[r] + tf[(size_t)tg * CH + m + 16];
  }
  __syncthreads();
  int tid = threadIdx.x;
  if (tid < 128) {                                 // per-token LN over 32 ch
    float sum = 0.f, ss = 0.f;
#pragma unroll
    for (int c = 0; c < CH; ++c) { float v = ldsF[tid * CH + c]; sum += v; ss += v * v; }
    float mean = sum * (1.f / CH);
    float var = ss * (1.f / CH) - mean * mean;
    float inv = rsqrtf(var + 1e-5f);
    int tg = blockIdx.x * 128 + tid;
    float* op = tokout + ((size_t)b * 16384 + tg) * CH;
#pragma unroll
    for (int c = 0; c < CH; ++c) op[c] = (ldsF[tid * CH + c] - mean) * inv;
  }
}

// ---------------- hash recon (LDS scatter-add) fused with final LN ---------
// grid = NB*NLEV, block = 1024, dynamic LDS = 64KB + reduction slots
__global__ void __launch_bounds__(1024) k_recon(const float* tokout, float* out) {
  extern __shared__ float tbl[];                 // 16384 table + 66 reduction
  float* pr = tbl + TDIM * 2;
  int bl = blockIdx.x, b = bl >> 4, l = bl & 15, tid = threadIdx.x;
  for (int i = tid; i < TDIM * 2; i += 1024) tbl[i] = 0.f;
  __syncthreads();
  int   rm = c_res[l] - 1;
  float sc = (float)rm * (1.0f / HRES);
  const float* tp = tokout + (size_t)b * 16384 * CH;
  for (int p = tid; p < HRES * HRES; p += 1024) {
    int x = p & 127, y = p >> 7;
    float fx = ((float)x + 0.5f) * sc, fy = ((float)y + 0.5f) * sc;
    float flx = floorf(fx), fly = floorf(fy);
    int xi0 = (int)flx, yi0 = (int)fly;
    float tx = fx - flx, ty = fy - fly;
    int xi1 = min(xi0 + 1, rm), yi1 = min(yi0 + 1, rm);
    float w00 = (1.f - tx) * (1.f - ty), w10 = tx * (1.f - ty);
    float w01 = (1.f - tx) * ty,         w11 = tx * ty;
    unsigned h0 = (unsigned)yi0 * 2654435761u, h1 = (unsigned)yi1 * 2654435761u;
    unsigned i00 = ((unsigned)xi0 ^ h0) & 8191u, i10 = ((unsigned)xi1 ^ h0) & 8191u;
    unsigned i01 = ((unsigned)xi0 ^ h1) & 8191u, i11 = ((unsigned)xi1 ^ h1) & 8191u;
    float f0 = tp[(size_t)p * CH + 2 * l];
    float f1 = tp[(size_t)p * CH + 2 * l + 1];
    atomicAdd(&tbl[2*i00],   f0 * w00); atomicAdd(&tbl[2*i00+1], f1 * w00);
    atomicAdd(&tbl[2*i10],   f0 * w10); atomicAdd(&tbl[2*i10+1], f1 * w10);
    atomicAdd(&tbl[2*i01],   f0 * w01); atomicAdd(&tbl[2*i01+1], f1 * w01);
    atomicAdd(&tbl[2*i11],   f0 * w11); atomicAdd(&tbl[2*i11+1], f1 * w11);
  }
  __syncthreads();
  float s = 0.f, ss = 0.f;                        // row LN over 16384
  for (int i = tid; i < TDIM * 2; i += 1024) { float v = tbl[i]; s += v; ss += v * v; }
  for (int m = 16; m >= 1; m >>= 1) { s += __shfl_xor(s, m, 32); ss += __shfl_xor(ss, m, 32); }
  if ((tid & 31) == 0) { pr[(tid >> 5) * 2] = s; pr[(tid >> 5) * 2 + 1] = ss; }
  __syncthreads();
  if (tid == 0) {
    float S = 0.f, SS = 0.f;
    for (int i = 0; i < 32; ++i) { S += pr[2 * i]; SS += pr[2 * i + 1]; }
    float mean = S * (1.f / 16384.f);
    float var = SS * (1.f / 16384.f) - mean * mean;
    pr[64] = mean; pr[65] = rsqrtf(var + 1e-5f);
  }
  __syncthreads();
  float mean = pr[64], inv = pr[65];
  float* o = out + (size_t)bl * (TDIM * 2);
  for (int i = tid; i < TDIM * 2; i += 1024) o[i] = (tbl[i] - mean) * inv;
}

// ---------------- launch ----------------
extern "C" void kernel_launch(void* const* d_in, const int* in_sizes, int n_in,
                              void* d_out, int out_size, void* d_ws, size_t ws_size,
                              hipStream_t stream) {
  (void)in_sizes; (void)n_in; (void)out_size; (void)ws_size;
  const float* inputs = (const float*)d_in[0];
  const float* s      = (const float*)d_in[1];
  const float* aw     = (const float*)d_in[2];
  const float* ab     = (const float*)d_in[3];
  const float* weight = (const float*)d_in[4];
  const float* bias   = (const float*)d_in[5];
  const float* kaw    = (const float*)d_in[6];
  const float* kab    = (const float*)d_in[7];
  const float* kwm    = (const float*)d_in[8];
  const float* oaw    = (const float*)d_in[9];
  const float* oab    = (const float*)d_in[10];
  const float* owm    = (const float*)d_in[11];

  char* wsp = (char*)d_ws;
  size_t off = 0;
  auto take = [&](size_t bytes) -> void* {
    void* p = wsp + off;
    off = (off + bytes + 255) & ~(size_t)255;
    return p;
  };
  float*    sst  = (float*)take((size_t)NB * CH * 4);
  float*    kst  = (float*)take((size_t)NB * CH * 4);
  float*    ost  = (float*)take((size_t)NB * CH * 4);
  _Float16* wT   = (_Float16*)take((size_t)NB * CH * 288 * 2);
  _Float16* kB   = (_Float16*)take((size_t)NB * CH * CH * 2);
  _Float16* oB   = (_Float16*)take((size_t)NB * CH * CH * 2);
  _Float16* xh   = (_Float16*)take((size_t)NB * CH * PADA * 2);
  float*    y    = (float*)take((size_t)NB * CH * RES * RES * 4);
  float*    tokf = (float*)take((size_t)NB * 16384 * CH * 4);
  _Float16* tokh = (_Float16*)take((size_t)NB * 16384 * CH * 2);
  float*    toko = (float*)take((size_t)NB * 16384 * CH * 4);

  long nz = (long)NB * CH * PADA / 2;              // f16 count / 2 -> u32 count
  k_zero32<<<(int)((nz + 255) / 256), 256, 0, stream>>>((unsigned int*)xh, nz);
  k_styles<<<3, 128, 0, stream>>>(s, aw, ab, kaw, kab, oaw, oab, sst, kst, ost);
  k_modprep<<<NB * CH, 320, 0, stream>>>(weight, kwm, owm, sst, kst, ost, wT, kB, oB);
  k_retrieve<<<NB * NLEV, 1024, TDIM * 2 * sizeof(float), stream>>>(inputs, xh);
  k_conv<<<dim3(512, NB), 256, 0, stream>>>(xh, wT, y);
  k_blur<<<8192, 256, 0, stream>>>(y, bias, tokf, tokh);
  k_mha<<<dim3(128, NB), 256, 0, stream>>>(tokf, tokh, kB, oB, toko);
  k_recon<<<NB * NLEV, 1024, (TDIM * 2 + 68) * sizeof(float), stream>>>(toko, (float*)d_out);
}
